// UniversalAttnModel_79293686218946
// MI455X (gfx1250) — matrix operand order repaired and checked by reading, not compiled, not run on hardware
//
#include <hip/hip_runtime.h>
#include <hip/hip_bf16.h>

// ---------------------------------------------------------------------------
// UniversalAttnModel for MI455X (gfx1250), wave32 + WMMA f16->f32 with
// async-to-LDS double buffering of the K stream (ASYNCcnt path).
//   B=64, J=32, N=4096, Q_DIM=130, K_DIM=128, V_DIM=130, H=4
// ---------------------------------------------------------------------------

typedef __attribute__((ext_vector_type(16))) _Float16 v16h;
typedef __attribute__((ext_vector_type(8)))  _Float16 v8h;
typedef __attribute__((ext_vector_type(2)))  _Float16 v2h;
typedef __attribute__((ext_vector_type(8)))  float    v8f;
typedef int v4i_t __attribute__((vector_size(4 * sizeof(int))));

#define BATCH  64
#define JDIM   32
#define NKEYS  4096
#define QDIM   130
#define KDIM   128
#define VDIM   130
#define HEADS  4
#define ROWS   (JDIM * HEADS)        // 128 query rows per batch
#define VPAD   144                   // V cols padded to 9 N-tiles of 16
#define NSPL   4                     // N-dimension splits
#define NB     32                    // keys per chunk
#define CHUNKS (NKEYS / NSPL / NB)   // 32

#if defined(__has_builtin)
#  if __has_builtin(__builtin_amdgcn_global_load_async_to_lds_b128)
#    define HAVE_ASYNC_LDS 1
#  else
#    define HAVE_ASYNC_LDS 0
#  endif
#else
#  define HAVE_ASYNC_LDS 0
#endif

#define GAS __attribute__((address_space(1)))
#define LAS __attribute__((address_space(3)))

// copy 16 bytes global -> LDS (async on CDNA5, sync float4 fallback)
static __device__ __forceinline__ void copy16_g2l(const void* g, void* l) {
#if HAVE_ASYNC_LDS
  __builtin_amdgcn_global_load_async_to_lds_b128((GAS v4i_t*)(g),
                                                 (LAS v4i_t*)(l), 0, 0);
#else
  *(float4*)l = *(const float4*)g;
#endif
}

static __device__ __forceinline__ void wait_async_zero() {
#if HAVE_ASYNC_LDS
#  if __has_builtin(__builtin_amdgcn_s_wait_asynccnt)
  __builtin_amdgcn_s_wait_asynccnt(0);
#  else
  asm volatile("s_wait_asynccnt 0x0" ::: "memory");
#  endif
#endif
}

static __device__ __forceinline__ v16h cat8(v8h lo, v8h hi) {
  return __builtin_shufflevector(lo, hi, 0,1,2,3,4,5,6,7,8,9,10,11,12,13,14,15);
}

// ---------------------------------------------------------------------------
// Kernel 1: query encoder.  relu(x@W1+b1) @ W2 + b2, scaled by 1/sqrt(128),
// written as f16 [b][row = j*4+head][k].  One block per (b,j), 64 threads.
// ---------------------------------------------------------------------------
__global__ void __launch_bounds__(64)
encoder_kernel(const float* __restrict__ jq,
               const float* __restrict__ W1, const float* __restrict__ b1,
               const float* __restrict__ W2, const float* __restrict__ b2,
               _Float16* __restrict__ qout) {
  const int bj = blockIdx.x;
  const int b  = bj / JDIM;
  const int j  = bj - b * JDIM;
  const int t  = threadIdx.x;

  __shared__ float x[QDIM];
  __shared__ float h[64];

  for (int c = t; c < QDIM; c += 64) x[c] = jq[(size_t)bj * QDIM + c];
  __syncthreads();

  float a = b1[t];
  for (int c = 0; c < QDIM; ++c) a = fmaf(x[c], W1[c * 64 + t], a);
  h[t] = fmaxf(a, 0.0f);
  __syncthreads();

  const float scl = 0.08838834764831845f;  // 1/sqrt(K_DIM)
  for (int i = 0; i < 8; ++i) {
    const int o = i * 64 + t;              // 0..511 = head*128 + k
    float a2 = b2[o];
    for (int c = 0; c < 64; ++c) a2 = fmaf(h[c], W2[c * 512 + o], a2);
    const int r   = j * HEADS + (o >> 7);
    const int col = o & (KDIM - 1);
    qout[((size_t)b * ROWS + r) * KDIM + col] = (_Float16)(a2 * scl);
  }
}

// ---------------------------------------------------------------------------
// Kernel 2: flash attention with WMMA + double-buffered data movement.
// grid = (NSPL, BATCH); block = 256 threads = 8 wave32s; wave w owns 16 rows.
// Per chunk: K fp32 async->LDS (double buffer), V register-staged -> f16
// transposed LDS, 8 score WMMAs + online softmax + 9 PV WMMAs per wave.
// ---------------------------------------------------------------------------
__global__ void __launch_bounds__(256)
attn_kernel(const _Float16* __restrict__ qin,
            const float* __restrict__ vk,
            const float* __restrict__ vv,
            const int*   __restrict__ mk,      // vertex_mask (nonzero = masked)
            float* __restrict__ pout,          // [B][NSPL][ROWS][VDIM]
            float* __restrict__ pm,            // [B][NSPL][ROWS]
            float* __restrict__ pl) {          // [B][NSPL][ROWS]
  const int split = blockIdx.x;
  const int b     = blockIdx.y;
  const int tid   = threadIdx.x;
  const int wave  = tid >> 5;
  const int lane  = tid & 31;
  const int l16   = lane & 15;
  const int h8    = lane >> 4;

  __shared__ __align__(16) float     Kf[2][NB][KDIM];     // 32 KB, fp32
  __shared__ __align__(16) _Float16  Vt[2][VPAD][NB];     // 18 KB, f16 transposed
  __shared__ __align__(16) _Float16  Plds[8][16][NB];     //  8 KB, per-wave P
  __shared__ float biasL[2][NB];

  const int nbase = split * (NKEYS / NSPL);

  // zero the V padding rows of both buffers (never rewritten)
  for (int idx = tid; idx < 2 * (VPAD - VDIM) * NB; idx += 256) {
    const int bb = idx / ((VPAD - VDIM) * NB);
    const int r  = idx - bb * (VPAD - VDIM) * NB;
    Vt[bb][VDIM + r / NB][r & (NB - 1)] = (_Float16)0.0f;
  }

  // ---- preload Q A-fragments (16 rows x 128 feats -> 4 frags, f16) ----
  const int mrow = wave * 16 + l16;
  const _Float16* qrow = qin + ((size_t)b * ROWS + mrow) * KDIM;
  v16h qfrag[4];
#pragma unroll
  for (int ks = 0; ks < 4; ++ks) {
    const int fb = ks * 32 + h8 * 16;      // A layout: e0-7, e8-15 at +8
    v8h lo = *(const v8h*)(qrow + fb);
    v8h hi = *(const v8h*)(qrow + fb + 8);
    qfrag[ks] = cat8(lo, hi);
  }

  // ---- softmax state + accumulators ----
  float m_s[8], l_s[8];
#pragma unroll
  for (int i = 0; i < 8; ++i) { m_s[i] = -3.0e38f; l_s[i] = 0.0f; }
  v8f acc[9];
#pragma unroll
  for (int t2 = 0; t2 < 9; ++t2)
#pragma unroll
    for (int i = 0; i < 8; ++i) acc[t2][i] = 0.0f;

  // ---------- staging helpers ----------
  v2h  vst[9];      // register-staged V chunk (f16 pairs)
  float breg = 0.0f;

  auto issueK = [&](int it, int bufn) {    // K chunk -> LDS buf (async b128)
    const float* Kg = vk + ((size_t)b * NKEYS + nbase + it * NB) * KDIM;
    float* kl = &Kf[bufn][0][0];
#pragma unroll
    for (int k = 0; k < (NB * KDIM / 4) / 256; ++k) {   // 4 x b128 per thread
      const int off = (tid + 256 * k) * 4;
      copy16_g2l(Kg + off, kl + off);
    }
  };
  auto loadV = [&](int it) {               // V chunk -> regs (cvt f16)
    const float* Vg = vv + ((size_t)b * NKEYS + nbase + it * NB) * VDIM;
#pragma unroll
    for (int k = 0; k < 9; ++k) {
      const int ia = tid + 256 * k;
      const int ib = tid + 256 * (k + 9);
      const float fa = (ia < NB * VDIM) ? Vg[ia] : 0.0f;
      const float fb = (ib < NB * VDIM) ? Vg[ib] : 0.0f;
      vst[k] = (v2h){(_Float16)fa, (_Float16)fb};
    }
  };
  auto storeV = [&](int bufn) {            // regs -> transposed f16 LDS
#pragma unroll
    for (int k = 0; k < 9; ++k) {
      const int ia = tid + 256 * k;
      const int ib = tid + 256 * (k + 9);
      if (ia < NB * VDIM) {
        const int key = ia / VDIM;
        Vt[bufn][ia - key * VDIM][key] = vst[k].x;
      }
      if (ib < NB * VDIM) {
        const int key = ib / VDIM;
        Vt[bufn][ib - key * VDIM][key] = vst[k].y;
      }
    }
  };
  auto loadB = [&](int it) {
    breg = (tid < NB && mk[(size_t)b * NKEYS + nbase + it * NB + tid]) ? -1.0e9f
                                                                       : 0.0f;
  };

  // ---------- prologue: stage chunk 0 ----------
  issueK(0, 0);
  loadV(0);
  loadB(0);

  for (int it = 0; it < CHUNKS; ++it) {
    const int cur = it & 1;

    // publish register-staged V/bias for this chunk
    storeV(cur);
    if (tid < NB) biasL[cur][tid] = breg;

    wait_async_zero();    // K async copies for buf[cur] complete
    __syncthreads();      // tile buf[cur] visible to everyone

    // stage next chunk while we compute (async K + V loads in flight)
    if (it + 1 < CHUNKS) {
      issueK(it + 1, 1 - cur);
      loadV(it + 1);
      loadB(it + 1);
    }

    // ---- scores: S = Q * K^T  (2 N-tiles x 4 K-steps of WMMA) ----
    v8f ct[2];
#pragma unroll
    for (int nt = 0; nt < 2; ++nt) {
      v8f c;
#pragma unroll
      for (int i = 0; i < 8; ++i) c[i] = 0.0f;
      const int key = nt * 16 + l16;               // B-frag: N = lane%16
#pragma unroll
      for (int ks = 0; ks < 4; ++ks) {
        const float* kp = &Kf[cur][key][ks * 32 + h8 * 16];
        const float4 f0 = ((const float4*)kp)[0];
        const float4 f1 = ((const float4*)kp)[1];
        const float4 f2 = ((const float4*)kp)[2];
        const float4 f3 = ((const float4*)kp)[3];
        v16h bf = {(_Float16)f0.x, (_Float16)f0.y, (_Float16)f0.z, (_Float16)f0.w,
                   (_Float16)f1.x, (_Float16)f1.y, (_Float16)f1.z, (_Float16)f1.w,
                   (_Float16)f2.x, (_Float16)f2.y, (_Float16)f2.z, (_Float16)f2.w,
                   (_Float16)f3.x, (_Float16)f3.y, (_Float16)f3.z, (_Float16)f3.w};
        c = __builtin_amdgcn_wmma_f32_16x16x32_f16(false, qfrag[ks], false, bf,
                                                   (short)0, c, false, false);
      }
      const float bb = biasL[cur][nt * 16 + l16];
#pragma unroll
      for (int i = 0; i < 8; ++i) c[i] += bb;
      ct[nt] = c;
    }

    // ---- online softmax (row groups live in 16-lane halves) ----
    float scale_i[8];
#pragma unroll
    for (int i = 0; i < 8; ++i) {
      float mx = fmaxf(ct[0][i], ct[1][i]);
#pragma unroll
      for (int off = 8; off >= 1; off >>= 1) mx = fmaxf(mx, __shfl_xor(mx, off, 32));
      const float mnew = fmaxf(m_s[i], mx);
      const float sc   = __expf(m_s[i] - mnew);
      float ps = 0.0f;
#pragma unroll
      for (int nt = 0; nt < 2; ++nt) {
        const float p = __expf(ct[nt][i] - mnew);
        ct[nt][i] = p;
        ps += p;
      }
#pragma unroll
      for (int off = 8; off >= 1; off >>= 1) ps += __shfl_xor(ps, off, 32);
      l_s[i] = l_s[i] * sc + ps;
      m_s[i] = mnew;
      scale_i[i] = sc;
    }
#pragma unroll
    for (int t2 = 0; t2 < 9; ++t2)
#pragma unroll
      for (int i = 0; i < 8; ++i) acc[t2][i] *= scale_i[i];

    // ---- re-stripe P (C layout) -> wave-private LDS -> A layout ----
#pragma unroll
    for (int nt = 0; nt < 2; ++nt)
#pragma unroll
      for (int i = 0; i < 8; ++i)
        Plds[wave][i + h8 * 8][nt * 16 + l16] = (_Float16)ct[nt][i];

    // ---- PV: acc += P * V  (1 K-step x 9 V-tiles of WMMA) ----
    {
      const int kb = h8 * 16;
      v8h alo = *(const v8h*)(&Plds[wave][l16][kb]);
      v8h ahi = *(const v8h*)(&Plds[wave][l16][kb + 8]);
      v16h af = cat8(alo, ahi);
#pragma unroll
      for (int t2 = 0; t2 < 9; ++t2) {
        const int vcol = t2 * 16 + l16;
        v8h blo = *(const v8h*)(&Vt[cur][vcol][h8 * 16]);
        v8h bhi = *(const v8h*)(&Vt[cur][vcol][h8 * 16 + 8]);
        v16h bf = cat8(blo, bhi);
        acc[t2] = __builtin_amdgcn_wmma_f32_16x16x32_f16(false, af, false, bf,
                                                         (short)0, acc[t2],
                                                         false, false);
      }
    }
  }

  // ---- write split partials ----
  const size_t pbase = ((size_t)b * NSPL + split) * ROWS;
  float* po = pout + pbase * VDIM;
#pragma unroll
  for (int t2 = 0; t2 < 9; ++t2)
#pragma unroll
    for (int i = 0; i < 8; ++i) {
      const int row = wave * 16 + i + h8 * 8;
      const int col = t2 * 16 + l16;
      if (col < VDIM) po[(size_t)row * VDIM + col] = acc[t2][i];
    }
  if (l16 == 0) {
#pragma unroll
    for (int i = 0; i < 8; ++i) {
      const int row = wave * 16 + i + h8 * 8;
      pm[pbase + row] = m_s[i];
      pl[pbase + row] = l_s[i];
    }
  }
}

// ---------------------------------------------------------------------------
// Kernel 3: combine N-splits (log-sum-exp merge) + output MLP 520->64->32->1.
// ---------------------------------------------------------------------------
__global__ void __launch_bounds__(64)
out_kernel(const float* __restrict__ pout, const float* __restrict__ pm,
           const float* __restrict__ pl,
           const float* __restrict__ W3, const float* __restrict__ b3,
           const float* __restrict__ W4, const float* __restrict__ b4,
           const float* __restrict__ W5, const float* __restrict__ b5,
           float* __restrict__ out) {
  const int bj = blockIdx.x;
  const int b  = bj / JDIM;
  const int j  = bj - b * JDIM;
  const int t  = threadIdx.x;

  __shared__ float feat[HEADS * VDIM];     // 520
  __shared__ float Dh[HEADS];
  __shared__ float Eh[HEADS][NSPL];
  __shared__ float h1[64], h2[32];

  if (t < HEADS) {
    const int row = j * HEADS + t;
    float M = -3.0e38f;
    for (int s = 0; s < NSPL; ++s)
      M = fmaxf(M, pm[((size_t)b * NSPL + s) * ROWS + row]);
    float den = 0.0f;
    for (int s = 0; s < NSPL; ++s) {
      const size_t idx = ((size_t)b * NSPL + s) * ROWS + row;
      const float e = __expf(pm[idx] - M);
      Eh[t][s] = e;
      den += e * pl[idx];
    }
    Dh[t] = den;
  }
  __syncthreads();

  for (int c = t; c < HEADS * VDIM; c += 64) {
    const int hh = c / VDIM;
    const int v  = c - hh * VDIM;
    const int row = j * HEADS + hh;
    float sacc = 0.0f;
    for (int s = 0; s < NSPL; ++s)
      sacc += Eh[hh][s] *
              pout[(((size_t)b * NSPL + s) * ROWS + row) * VDIM + v];
    feat[c] = sacc / Dh[hh];
  }
  __syncthreads();

  float a = b3[t];
  for (int c = 0; c < HEADS * VDIM; ++c) a = fmaf(feat[c], W3[c * 64 + t], a);
  h1[t] = fmaxf(a, 0.0f);
  __syncthreads();

  if (t < 32) {
    float a2 = b4[t];
    for (int c = 0; c < 64; ++c) a2 = fmaf(h1[c], W4[c * 32 + t], a2);
    h2[t] = fmaxf(a2, 0.0f);
  }
  __syncthreads();

  if (t == 0) {
    float a3 = b5[0];
    for (int c = 0; c < 32; ++c) a3 = fmaf(h2[c], W5[c], a3);
    out[bj] = a3;
  }
}

// ---------------------------------------------------------------------------
extern "C" void kernel_launch(void* const* d_in, const int* in_sizes, int n_in,
                              void* d_out, int out_size, void* d_ws, size_t ws_size,
                              hipStream_t stream) {
  const float* jq  = (const float*)d_in[0];
  const float* vk  = (const float*)d_in[1];
  const float* vv  = (const float*)d_in[2];
  const int*   msk = (const int*)  d_in[3];
  const float* W1  = (const float*)d_in[4];
  const float* b1  = (const float*)d_in[5];
  const float* W2  = (const float*)d_in[6];
  const float* b2  = (const float*)d_in[7];
  const float* W3  = (const float*)d_in[8];
  const float* b3  = (const float*)d_in[9];
  const float* W4  = (const float*)d_in[10];
  const float* b4  = (const float*)d_in[11];
  const float* W5  = (const float*)d_in[12];
  const float* b5  = (const float*)d_in[13];
  float* out = (float*)d_out;

  // workspace layout (~19.4 MB total)
  char* w = (char*)d_ws;
  _Float16* qf16 = (_Float16*)w;
  const size_t qbytes = (size_t)BATCH * ROWS * KDIM * sizeof(_Float16);
  float* pout = (float*)(w + qbytes);
  float* pm   = pout + (size_t)BATCH * NSPL * ROWS * VDIM;
  float* pl   = pm   + (size_t)BATCH * NSPL * ROWS;
  (void)ws_size; (void)n_in; (void)in_sizes; (void)out_size;

  encoder_kernel<<<BATCH * JDIM, 64, 0, stream>>>(jq, W1, b1, W2, b2, qf16);
  attn_kernel<<<dim3(NSPL, BATCH), 256, 0, stream>>>(qf16, vk, vv, msk,
                                                     pout, pm, pl);
  out_kernel<<<BATCH * JDIM, 64, 0, stream>>>(pout, pm, pl,
                                              W3, b3, W4, b4, W5, b5, out);
}
